// FastRowRNN_26860725469730
// MI455X (gfx1250) — matrix operand order repaired and checked by reading, not compiled, not run on hardware
//
#include <hip/hip_runtime.h>
#include <hip/hip_bf16.h>
#include <math.h>

// ---------------------------------------------------------------------------
// FastRowRNN on gfx1250: wavefront-pipelined 3-layer GRU with WMMA bf16 GEMMs
//   B=16, ROWS=128, T=128, E=64 (+1 rowfeat -> Kpad 96), H=32, gates=96, V=258
// ---------------------------------------------------------------------------

typedef __attribute__((ext_vector_type(16))) __bf16 v16bf;
typedef __attribute__((ext_vector_type(8)))  __bf16 v8bf;
typedef __attribute__((ext_vector_type(8)))  float  v8f;

#define NROWS 128
#define TT    128
#define NB    16
#define EPAD  96
#define HH    32
#define NV    258

// packed weight tile table (each tile = 512 bf16 = 32 lanes x 16)
#define TILE_AIH0 0    // 6 mt x 3 kt
#define TILE_AHH0 18   // 6
#define TILE_AIH1 24   // 6
#define TILE_AHH1 30   // 6
#define TILE_AIH2 36   // 6
#define TILE_AHH2 42   // 6
#define TILE_H2E  48   // 5
#define TILE_OUT  53   // 17
#define NTILES    70

// workspace layout (bytes)
#define OFF_PACK 0                          // 71680 B
#define OFF_XIN  (131072)                   // 128*128*16*96*2 = 50331648 B
#define OFF_OUT  (131072 + 50331648)        // 128*128*16*32*2 = 16777216 B
#define OFF_FLAG (131072 + 50331648 + 16777216) // 128*128*4 B

__device__ __forceinline__ v8f wmma_bf16(v16bf a, v16bf b, v8f c) {
    return __builtin_amdgcn_wmma_f32_16x16x32_bf16(false, a, false, b,
                                                   (short)0, c, false, false);
}

__device__ __forceinline__ float fast_sigmoid(float x) {
    return 1.0f / (1.0f + __expf(-x));
}
__device__ __forceinline__ float fast_tanh(float x) {
    return 2.0f / (1.0f + __expf(-2.0f * x)) - 1.0f;
}

// B fragment: activations stored batch-major [n][kpad] bf16 in LDS/global.
// lane: n = lane%16, k = kstep*32 + (lane>=16 ? 16 : 0) .. +15 (contiguous)
__device__ __forceinline__ v16bf load_bfrag(const __bf16* base, int lane,
                                            int kpad, int kstep) {
    int n  = lane & 15;
    int kb = kstep * 32 + ((lane >> 4) << 4);
    return *(const v16bf*)(base + n * kpad + kb);
}

// D tile (f32 16x16) -> row-major [16][16] f32 in LDS
__device__ __forceinline__ void store_gates(float* dst, int lane, v8f c) {
    int n  = lane & 15;
    int hi = (lane >> 4) * 8;
#pragma unroll
    for (int r = 0; r < 8; ++r) dst[(hi + r) * 16 + n] = c[r];
}

// ---------------------------------------------------------------------------
// Stage 0a: pack weight matrices into WMMA A-fragment layout (bf16).
// A layout: m = mt*16 + lane%16 ; k = kt*32 + (lane>=16?16:0) + s
//                                 (contiguous, slot s = 0..15)
// ---------------------------------------------------------------------------
__global__ void pack_weights_kernel(const float* __restrict__ wih0,
                                    const float* __restrict__ whh0,
                                    const float* __restrict__ wih1,
                                    const float* __restrict__ whh1,
                                    const float* __restrict__ wih2,
                                    const float* __restrict__ whh2,
                                    const float* __restrict__ wh2e,
                                    const float* __restrict__ wout,
                                    __bf16* __restrict__ dst) {
    int gid  = blockIdx.x * blockDim.x + threadIdx.x;
    int tile = gid >> 5;
    int lane = gid & 31;
    if (tile >= NTILES) return;

    const float* src; int M, K, nkt, lt;
    if      (tile < TILE_AHH0) { src = wih0; M = 96;  K = 65; nkt = 3; lt = tile; }
    else if (tile < TILE_AIH1) { src = whh0; M = 96;  K = 32; nkt = 1; lt = tile - TILE_AHH0; }
    else if (tile < TILE_AHH1) { src = wih1; M = 96;  K = 32; nkt = 1; lt = tile - TILE_AIH1; }
    else if (tile < TILE_AIH2) { src = whh1; M = 96;  K = 32; nkt = 1; lt = tile - TILE_AHH1; }
    else if (tile < TILE_AHH2) { src = wih2; M = 96;  K = 32; nkt = 1; lt = tile - TILE_AIH2; }
    else if (tile < TILE_H2E)  { src = whh2; M = 96;  K = 32; nkt = 1; lt = tile - TILE_AHH2; }
    else if (tile < TILE_OUT)  { src = wh2e; M = 65;  K = 32; nkt = 1; lt = tile - TILE_H2E; }
    else                       { src = wout; M = 258; K = 32; nkt = 1; lt = tile - TILE_OUT; }

    int mt = lt / nkt, kt = lt % nkt;
    int m  = mt * 16 + (lane & 15);
    int hi = (lane >> 4) ? 8 : 0;
#pragma unroll
    for (int s = 0; s < 16; ++s) {
        int j = s >> 1, p = s & 1;
        int k = kt * 32 + (hi << 1) + (j << 1) + p;
        float v = (m < M && k < K) ? src[m * K + k] : 0.0f;
        dst[tile * 512 + lane * 16 + s] = (__bf16)v;
    }
}

// ---------------------------------------------------------------------------
// Stage 0b: embedding gather + row feature, zero-padded to EPAD, bf16,
//           stored batch-major xin[row][t][b][0..95]
// ---------------------------------------------------------------------------
__global__ void embed_kernel(const int* __restrict__ x,
                             const float* __restrict__ tab,
                             __bf16* __restrict__ xin) {
    size_t idx = (size_t)blockIdx.x * blockDim.x + threadIdx.x;
    const size_t N = (size_t)NROWS * TT * NB * EPAD;
    if (idx >= N) return;
    int e   = (int)(idx % EPAD);
    size_t q = idx / EPAD;
    int b   = (int)(q % NB);  q /= NB;
    int t   = (int)(q % TT);
    int row = (int)(q / TT);
    float v;
    if (e < 64) {
        int xi = x[((size_t)b * NROWS + row) * TT + t];   // x[b][0][row][t]
        v = tab[(size_t)xi * 64 + e];
    } else if (e == 64) {
        v = (float)row / (float)NROWS * 2.0f - 1.0f;
    } else {
        v = 0.0f;
    }
    xin[idx] = (__bf16)v;
}

// ---------------------------------------------------------------------------
// Stage 1: persistent wavefront GRU. One block per row, 8 wave32s.
// ---------------------------------------------------------------------------
__global__ __launch_bounds__(256, 1) void gru_wave_kernel(
    const __bf16* __restrict__ xin, __bf16* __restrict__ outg,
    unsigned int* __restrict__ flags, const __bf16* __restrict__ packW,
    const float* __restrict__ bih0, const float* __restrict__ bhh0,
    const float* __restrict__ bih1, const float* __restrict__ bhh1,
    const float* __restrict__ bih2, const float* __restrict__ bhh2,
    const float* __restrict__ bh2e) {
    const int row = blockIdx.x;
    const int tid = threadIdx.x, lane = tid & 31, w = tid >> 5;
    const int n = lane & 15, hi8 = (lane >> 4) * 8, kbB = (lane >> 4) * 16;

    __shared__ __attribute__((aligned(32))) __bf16 actA[NB * EPAD];   // l0 input
    __shared__ __attribute__((aligned(32))) __bf16 actB[NB * HH];     // l1/l2 input
    __shared__ __attribute__((aligned(32))) __bf16 hb[3][NB * HH];    // h (bf16)
    __shared__ __attribute__((aligned(32))) float hprev[3][NB * HH];  // h (f32)
    __shared__ __attribute__((aligned(32))) float rzb[64 * 16];
    __shared__ __attribute__((aligned(32))) float innb[32 * 16];
    __shared__ __attribute__((aligned(32))) float hnb[32 * 16];
    __shared__ __attribute__((aligned(32))) float brz[3][64];
    __shared__ __attribute__((aligned(32))) float binb[3][32];
    __shared__ __attribute__((aligned(32))) float bhnb[3][32];
    __shared__ __attribute__((aligned(32))) float bhe[80];            // padded

    for (int i = tid; i < NB * EPAD; i += 256) actA[i] = (__bf16)0.0f;
    for (int i = tid; i < NB * HH; i += 256) {
        actB[i] = (__bf16)0.0f;
#pragma unroll
        for (int l = 0; l < 3; ++l) { hb[l][i] = (__bf16)0.0f; hprev[l][i] = 0.0f; }
    }
    for (int i = tid; i < 64; i += 256) {
        brz[0][i] = bih0[i] + bhh0[i];
        brz[1][i] = bih1[i] + bhh1[i];
        brz[2][i] = bih2[i] + bhh2[i];
    }
    for (int i = tid; i < 32; i += 256) {
        binb[0][i] = bih0[64 + i]; binb[1][i] = bih1[64 + i]; binb[2][i] = bih2[64 + i];
        bhnb[0][i] = bhh0[64 + i]; bhnb[1][i] = bhh1[64 + i]; bhnb[2][i] = bhh2[64 + i];
    }
    for (int i = tid; i < 80; i += 256) bhe[i] = (i < 65) ? bh2e[i] : 0.0f;

    // -------- register-resident A fragments (persist across all t) --------
    auto ldA = [&](int tile) -> v16bf {
        return *(const v16bf*)(packW + tile * 512 + lane * 16);
    };
    v16bf f_h2e{}, f_ih0a{}, f_ih0b{}, f_ih0c{}, f_hh0{}, f_ih1{}, f_hh1{}, f_ih2{}, f_hh2{};
    if (w < 5) f_h2e = ldA(TILE_H2E + w);
    if (w < 4) {
        f_ih0a = ldA(TILE_AIH0 + w * 3 + 0);
        f_ih0b = ldA(TILE_AIH0 + w * 3 + 1);
        f_ih0c = ldA(TILE_AIH0 + w * 3 + 2);
        f_hh0  = ldA(TILE_AHH0 + w);
        f_ih1  = ldA(TILE_AIH1 + w);  f_hh1 = ldA(TILE_AHH1 + w);
        f_ih2  = ldA(TILE_AIH2 + w);  f_hh2 = ldA(TILE_AHH2 + w);
    } else if (w < 6) {               // inn tiles: gate rows 64..95
        f_ih0a = ldA(TILE_AIH0 + w * 3 + 0);
        f_ih0b = ldA(TILE_AIH0 + w * 3 + 1);
        f_ih0c = ldA(TILE_AIH0 + w * 3 + 2);
        f_ih1  = ldA(TILE_AIH1 + w);
        f_ih2  = ldA(TILE_AIH2 + w);
    } else {                          // hn tiles: gate rows 64..95
        int mt = w - 2;
        f_hh0 = ldA(TILE_AHH0 + mt);
        f_hh1 = ldA(TILE_AHH1 + mt);
        f_hh2 = ldA(TILE_AHH2 + mt);
    }
    __syncthreads();

    for (int t = 0; t < TT; ++t) {
        const int rt = row * TT + t;
        if (row > 0) {
            if (tid == 0) {
                while (__hip_atomic_load(&flags[rt - TT], __ATOMIC_ACQUIRE,
                                         __HIP_MEMORY_SCOPE_AGENT) == 0u)
                    __builtin_amdgcn_s_sleep(1);
            }
            __syncthreads();
        }
        // ---- comb = xin + W_h2e @ prev_out + b_h2e  (waves 0..4) ----
        // Branch-free: A-tile rows >=65 are packed zero, xin cols >=65 zero,
        // bhe padded zero -> the padded region computes exactly 0.
        if (w < 5) {
            v8f c = {0.f, 0.f, 0.f, 0.f, 0.f, 0.f, 0.f, 0.f};
            if (row > 0) {
                v16bf bf = *(const v16bf*)(outg + ((size_t)(rt - TT) * NB + n) * HH + kbB);
                c = wmma_bf16(f_h2e, bf, c);
            }
            const int m0 = w * 16 + hi8;                       // 0..79, 16B aligned
            v8bf xv = *(const v8bf*)(xin + (size_t)rt * NB * EPAD + n * EPAD + m0);
            v8f  bv = *(const v8f*)(&bhe[m0]);
            v8bf ov;
#pragma unroll
            for (int r = 0; r < 8; ++r)
                ov[r] = (__bf16)(c[r] + bv[r] + (float)xv[r]);
            *(v8bf*)(actA + n * EPAD + m0) = ov;
            if (t < TT - 1)
                __builtin_prefetch(xin + (size_t)(rt + 1) * NB * EPAD + n * EPAD + m0, 0, 1);
        }
        __syncthreads();

        // ---- 3 GRU layers ----
#pragma unroll
        for (int l = 0; l < 3; ++l) {
            v8f c = {0.f, 0.f, 0.f, 0.f, 0.f, 0.f, 0.f, 0.f};
            if (w < 4) {                       // merged r,z gate rows (w*16..)
                if (l == 0) {
                    c = wmma_bf16(f_ih0a, load_bfrag(actA, lane, EPAD, 0), c);
                    c = wmma_bf16(f_ih0b, load_bfrag(actA, lane, EPAD, 1), c);
                    c = wmma_bf16(f_ih0c, load_bfrag(actA, lane, EPAD, 2), c);
                    c = wmma_bf16(f_hh0,  load_bfrag(hb[0], lane, HH, 0), c);
                } else if (l == 1) {
                    c = wmma_bf16(f_ih1, load_bfrag(actB, lane, HH, 0), c);
                    c = wmma_bf16(f_hh1, load_bfrag(hb[1], lane, HH, 0), c);
                } else {
                    c = wmma_bf16(f_ih2, load_bfrag(actB, lane, HH, 0), c);
                    c = wmma_bf16(f_hh2, load_bfrag(hb[2], lane, HH, 0), c);
                }
                store_gates(rzb + w * 256, lane, c);
            } else if (w < 6) {                // inn
                if (l == 0) {
                    c = wmma_bf16(f_ih0a, load_bfrag(actA, lane, EPAD, 0), c);
                    c = wmma_bf16(f_ih0b, load_bfrag(actA, lane, EPAD, 1), c);
                    c = wmma_bf16(f_ih0c, load_bfrag(actA, lane, EPAD, 2), c);
                } else if (l == 1) {
                    c = wmma_bf16(f_ih1, load_bfrag(actB, lane, HH, 0), c);
                } else {
                    c = wmma_bf16(f_ih2, load_bfrag(actB, lane, HH, 0), c);
                }
                store_gates(innb + (w - 4) * 256, lane, c);
            } else {                           // hn
                if (l == 0)      c = wmma_bf16(f_hh0, load_bfrag(hb[0], lane, HH, 0), c);
                else if (l == 1) c = wmma_bf16(f_hh1, load_bfrag(hb[1], lane, HH, 0), c);
                else             c = wmma_bf16(f_hh2, load_bfrag(hb[2], lane, HH, 0), c);
                store_gates(hnb + (w - 6) * 256, lane, c);
            }
            __syncthreads();

            // elementwise gate update: lane owns 8 consecutive h for one b ->
            // hprev/hb/actB/outg accesses are single vector ops.
            if (w < 2) {
                const int b  = lane & 15;
                const int h8 = (w * 2 + (lane >> 4)) * 8;      // 0,8,16,24
                float g_r[8], g_z[8], g_in[8], g_hn[8];
#pragma unroll
                for (int i = 0; i < 8; ++i) {
                    int h = h8 + i;
                    g_r[i]  = rzb[h * 16 + b];
                    g_z[i]  = rzb[(32 + h) * 16 + b];
                    g_in[i] = innb[h * 16 + b];
                    g_hn[i] = hnb[h * 16 + b];
                }
                v8f brv = *(const v8f*)(&brz[l][h8]);
                v8f bzv = *(const v8f*)(&brz[l][32 + h8]);
                v8f biv = *(const v8f*)(&binb[l][h8]);
                v8f bhv = *(const v8f*)(&bhnb[l][h8]);
                v8f hold = *(const v8f*)(&hprev[l][b * HH + h8]);
                v8f hnew;
                v8bf h16v;
#pragma unroll
                for (int i = 0; i < 8; ++i) {
                    float rr = fast_sigmoid(g_r[i] + brv[i]);
                    float zz = fast_sigmoid(g_z[i] + bzv[i]);
                    float nn = fast_tanh(g_in[i] + biv[i] + rr * (g_hn[i] + bhv[i]));
                    float hv = (1.0f - zz) * nn + zz * hold[i];
                    hnew[i] = hv;
                    h16v[i] = (__bf16)hv;
                }
                *(v8f*)(&hprev[l][b * HH + h8]) = hnew;
                *(v8bf*)(&hb[l][b * HH + h8])   = h16v;
                if (l < 2) {
                    *(v8bf*)(&actB[b * HH + h8]) = h16v;
                } else {
                    *(v8bf*)(outg + ((size_t)rt * NB + b) * HH + h8) = h16v;
                    __threadfence();           // release outg stores (agent scope)
                }
            }
            __syncthreads();
        }
        if (tid == 0)
            __hip_atomic_store(&flags[rt], 1u, __ATOMIC_RELEASE,
                               __HIP_MEMORY_SCOPE_AGENT);
    }
}

// ---------------------------------------------------------------------------
// Stage 2: output projection  pred[b][row][t][v] = out[rt][b] . w_out[v] + b_out
// ---------------------------------------------------------------------------
__global__ __launch_bounds__(256) void proj_kernel(
    const __bf16* __restrict__ outg, const __bf16* __restrict__ packW,
    const float* __restrict__ b_out, float* __restrict__ pred) {
    int rt = blockIdx.x;
    int tid = threadIdx.x, lane = tid & 31, w = tid >> 5;
    int row = rt >> 7, t = rt & 127;
    int n = lane & 15, hi = lane >> 4;

    v16bf bfrag = *(const v16bf*)(outg + ((size_t)rt * NB + n) * HH + hi * 16);
    float* base = pred + (((size_t)n * NROWS + row) * TT + t) * NV;

    // full tiles 0..15 (waves 0..7 take mt = w and mt = w+8): unconditional
#pragma unroll
    for (int it = 0; it < 2; ++it) {
        int mt = w + it * 8;
        v16bf a = *(const v16bf*)(packW + (TILE_OUT + mt) * 512 + lane * 16);
        v8f c = {0.f, 0.f, 0.f, 0.f, 0.f, 0.f, 0.f, 0.f};
        c = wmma_bf16(a, bfrag, c);
        int v0 = mt * 16 + hi * 8;                 // multiple of 8 -> 8B aligned
        v8f bo = *(const v8f*)(b_out + v0);
        float* dp = base + v0;
#pragma unroll
        for (int i = 0; i < 4; ++i) {
            float2 pr = make_float2(c[2 * i] + bo[2 * i], c[2 * i + 1] + bo[2 * i + 1]);
            *(float2*)(dp + 2 * i) = pr;
        }
    }
    // ragged tile 16 (v = 256..271, only 256/257 valid), wave 0 only
    if (w == 0) {
        v16bf a = *(const v16bf*)(packW + (TILE_OUT + 16) * 512 + lane * 16);
        v8f c = {0.f, 0.f, 0.f, 0.f, 0.f, 0.f, 0.f, 0.f};
        c = wmma_bf16(a, bfrag, c);
        if (hi == 0) {                             // rows 256,257 live in lanes 0-15
            float2 pr = make_float2(c[0] + b_out[256], c[1] + b_out[257]);
            *(float2*)(base + 256) = pr;
        }
    }
}

// ---------------------------------------------------------------------------
extern "C" void kernel_launch(void* const* d_in, const int* in_sizes, int n_in,
                              void* d_out, int out_size, void* d_ws, size_t ws_size,
                              hipStream_t stream) {
    (void)in_sizes; (void)n_in; (void)out_size; (void)ws_size;
    const int*   x    = (const int*)d_in[0];
    const float* tab  = (const float*)d_in[1];
    const float* wih0 = (const float*)d_in[2];
    const float* whh0 = (const float*)d_in[3];
    const float* bih0 = (const float*)d_in[4];
    const float* bhh0 = (const float*)d_in[5];
    const float* wih1 = (const float*)d_in[6];
    const float* whh1 = (const float*)d_in[7];
    const float* bih1 = (const float*)d_in[8];
    const float* bhh1 = (const float*)d_in[9];
    const float* wih2 = (const float*)d_in[10];
    const float* whh2 = (const float*)d_in[11];
    const float* bih2 = (const float*)d_in[12];
    const float* bhh2 = (const float*)d_in[13];
    const float* wh2e = (const float*)d_in[14];
    const float* bh2e = (const float*)d_in[15];
    const float* wout = (const float*)d_in[16];
    const float* bout = (const float*)d_in[17];

    char* ws = (char*)d_ws;
    __bf16*       packW = (__bf16*)(ws + OFF_PACK);
    __bf16*       xin   = (__bf16*)(ws + OFF_XIN);
    __bf16*       outg  = (__bf16*)(ws + OFF_OUT);
    unsigned int* flags = (unsigned int*)(ws + OFF_FLAG);

    hipMemsetAsync(flags, 0, NROWS * TT * sizeof(unsigned int), stream);

    pack_weights_kernel<<<(NTILES * 32 + 255) / 256, 256, 0, stream>>>(
        wih0, whh0, wih1, whh1, wih2, whh2, wh2e, wout, packW);

    size_t nEmb = (size_t)NROWS * TT * NB * EPAD;
    embed_kernel<<<(unsigned)((nEmb + 255) / 256), 256, 0, stream>>>(x, tab, xin);

    gru_wave_kernel<<<NROWS, 256, 0, stream>>>(xin, outg, flags, packW,
                                               bih0, bhh0, bih1, bhh1,
                                               bih2, bhh2, bh2e);

    proj_kernel<<<NROWS * TT, 256, 0, stream>>>(outg, packW, bout, (float*)d_out);
}
